// DifferentiableRenderer_10831907520799
// MI455X (gfx1250) — matrix operand order repaired and checked by reading, not compiled, not run on hardware
//
#include <hip/hip_runtime.h>
#include <hip/hip_bf16.h>
#include <hip/hip_fp16.h>

typedef __attribute__((ext_vector_type(16))) _Float16 v16h;
typedef __attribute__((ext_vector_type(8)))  float    v8f;

#define HH 128
#define WW 128
#define HWPIX (HH * WW)
#define NG 65536
#define NBLK (NG / 32)       // 2048 blocks of 32 gaussians
#define EPS 1e-8f
#define LOG2E 1.4426950408889634f

// ---------------------------------------------------------------------------
// Prep 1: per-camera projection, pre-shuffled into WMMA A-matrix lane/K order.
// pa[b*65536 + blk*32 + h*16 + j] = (proj_x, proj_y, -0.5*log2e/var, opacity)
// for gaussian g = blk*32 + K_A(h, j), where
//   K_A(0,j) = j                               -> lanes 0..15
//   K_A(1,j) = j+16                            -> lanes 16..31
// ---------------------------------------------------------------------------
__global__ void prep_params(const float* __restrict__ pos,
                            const float* __restrict__ scales,
                            const float* __restrict__ opac,
                            const float* __restrict__ qvec,
                            const float* __restrict__ tvec,
                            const float* __restrict__ fxp,
                            const float* __restrict__ fyp,
                            const float* __restrict__ cxp,
                            const float* __restrict__ cyp,
                            float4* __restrict__ pa)
{
    int tid    = blockIdx.x * blockDim.x + threadIdx.x;   // 0 .. 131071
    int b      = tid >> 16;
    int slot   = tid & 65535;
    int blk    = slot >> 5;
    int within = slot & 31;
    int h      = within >> 4;
    int j      = within & 15;
    int K      = h * 16 + j;
    int g      = blk * 32 + K;

    // rotation from quaternion
    float qw = qvec[b * 4 + 0], qx = qvec[b * 4 + 1];
    float qy = qvec[b * 4 + 2], qz = qvec[b * 4 + 3];
    float qn = rsqrtf(qw * qw + qx * qx + qy * qy + qz * qz);
    qw *= qn; qx *= qn; qy *= qn; qz *= qn;
    float r00 = 1.f - 2.f * (qy * qy + qz * qz), r01 = 2.f * (qx * qy - qz * qw), r02 = 2.f * (qx * qz + qy * qw);
    float r10 = 2.f * (qx * qy + qz * qw), r11 = 1.f - 2.f * (qx * qx + qz * qz), r12 = 2.f * (qy * qz - qx * qw);
    float r20 = 2.f * (qx * qz - qy * qw), r21 = 2.f * (qy * qz + qx * qw), r22 = 1.f - 2.f * (qx * qx + qy * qy);

    float px = pos[g * 3 + 0], py = pos[g * 3 + 1], pz = pos[g * 3 + 2];
    float camx = r00 * px + r01 * py + r02 * pz + tvec[b * 3 + 0];
    float camy = r10 * px + r11 * py + r12 * pz + tvec[b * 3 + 1];
    float camz = r20 * px + r21 * py + r22 * pz + tvec[b * 3 + 2];

    float invz  = 1.0f / camz;
    float projx = camx * invz * (*fxp) + (*cxp);
    float projy = camy * invz * (*fyp) + (*cyp);

    float s  = scales[g];
    float cc = -(0.5f * LOG2E) / (s * s);   // exp(-0.5*d2/var) == exp2(cc*d2)
    pa[tid]  = make_float4(projx, projy, cc, opac[g]);
}

// ---------------------------------------------------------------------------
// Prep 2: pack colors into WMMA B-matrix (32x16 f16) register layout.
// Lane n = column (n & 15); lanes 0-15 hold K=0..15, lanes 16-31 hold K=16..31.
// Columns: 0..2 = r,g,b ; 3 = 1.0 (density) ; 4..15 = 0.
// ---------------------------------------------------------------------------
__global__ void prep_btile(const float* __restrict__ colors, v16h* __restrict__ bt)
{
    int tid   = blockIdx.x * blockDim.x + threadIdx.x;    // 0 .. 65535
    int blk   = tid >> 5;
    int lane  = tid & 31;
    int col   = lane & 15;
    int kbase = (lane < 16) ? 0 : 16;
    v16h v;
#pragma unroll
    for (int j = 0; j < 16; ++j) {
        int   g   = blk * 32 + kbase + j;
        float val = (col < 3) ? colors[g * 3 + col] : ((col == 3) ? 1.0f : 0.0f);
        v[j] = (_Float16)val;
    }
    bt[tid] = v;
}

// ---------------------------------------------------------------------------
// Main: each wave owns 2 adjacent strips of 16 pixels (same image row) for one
// camera and sweeps all 2048 gaussian blocks. Strip 1 reuses strip 0's
// exponent: d2_1 = d2_0 + 32*dx + 256  ->  e1 = fma(P.z, fma(dx,32,256), e0).
// Per block: two f16 A tiles via v_exp_f32 -> two v_wmma_f32_16x16x32_f16.
// ---------------------------------------------------------------------------
__global__ void __launch_bounds__(128)
render_main(const float4* __restrict__ pa, const v16h* __restrict__ bt,
            float* __restrict__ out)
{
    __shared__ float sm[4][2][16][4];   // [wave][strip][pixel][col r,g,b,den]

    int lane  = threadIdx.x & 31;
    int wv    = threadIdx.x >> 5;
    int wglob = blockIdx.x * 4 + wv;    // 0 .. 1023
    int b     = wglob >> 9;             // camera
    int sIdx  = wglob & 511;
    int s0    = sIdx * 2, s1 = s0 + 1;  // two adjacent strips, same row

    int m  = lane & 15;                 // A-matrix row = pixel within strip
    int p0 = s0 * 16 + m;
    float x0 = (float)(p0 & (WW - 1));  // x1 = x0 + 16, y1 = y0
    float y0 = (float)(p0 >> 7);
    int half = lane >> 4;

    v8f c0 = {}, c1 = {};
    const float4* paCam = pa + (size_t)b * 65536;

    for (int blk = 0; blk < NBLK; ++blk) {
        v16h btv = bt[blk * 32 + lane];                       // B tile, 32B/lane
        const float4* pp = paCam + blk * 32 + half * 16;      // A-order params
        if (blk + 1 < NBLK) {                                 // warm L0 for next
            __builtin_prefetch(pp + 32, 0, 1);                //  block's params
            __builtin_prefetch(&bt[(blk + 1) * 32 + lane], 0, 1);
        }
        v16h a0, a1;
#pragma unroll
        for (int j = 0; j < 16; ++j) {
            float4 P  = pp[j];                                // px,py,c,op
            float dx  = x0 - P.x, dy = y0 - P.y;
            float e0  = P.z * fmaf(dy, dy, dx * dx);
            float w0  = P.w * __builtin_amdgcn_exp2f(e0);
            float e1  = fmaf(P.z, fmaf(dx, 32.0f, 256.0f), e0);
            float w1  = P.w * __builtin_amdgcn_exp2f(e1);
            a0[j] = (_Float16)w0;
            a1[j] = (_Float16)w1;
        }
        c0 = __builtin_amdgcn_wmma_f32_16x16x32_f16(false, a0, false, btv,
                                                    (short)0, c0, false, false);
        c1 = __builtin_amdgcn_wmma_f32_16x16x32_f16(false, a1, false, btv,
                                                    (short)0, c1, false, false);
    }

    // ---- epilogue: C layout -> LDS -> normalized (B,3,H,W) output ----
    int col = lane & 15;
    int mb  = half * 8;                 // lanes 16-31 hold rows M=8..15
    if (col < 4) {
#pragma unroll
        for (int j = 0; j < 8; ++j) {
            sm[wv][0][mb + j][col] = c0[j];
            sm[wv][1][mb + j][col] = c1[j];
        }
    }
    __syncthreads();

#pragma unroll
    for (int t = 0; t < 3; ++t) {
        int idx   = t * 32 + lane;      // 0..95 : 2 strips * 16 px * 3 ch
        int strip = idx / 48;
        int r     = idx % 48;
        int ch    = r >> 4;
        int px    = r & 15;
        float num = sm[wv][strip][px][ch];
        float den = sm[wv][strip][px][3] + 32.0f * EPS;  // +EPS per 2048-chunk
        den = fmaxf(den, EPS);
        int s = strip ? s1 : s0;
        out[(b * 3 + ch) * HWPIX + s * 16 + px] = num / den;
    }
}

// ---------------------------------------------------------------------------
extern "C" void kernel_launch(void* const* d_in, const int* in_sizes, int n_in,
                              void* d_out, int out_size, void* d_ws, size_t ws_size,
                              hipStream_t stream)
{
    const float* positions = (const float*)d_in[0];
    const float* colors    = (const float*)d_in[1];
    const float* opac      = (const float*)d_in[2];
    const float* scales    = (const float*)d_in[3];
    const float* qvec      = (const float*)d_in[4];
    const float* tvec      = (const float*)d_in[5];
    const float* fx        = (const float*)d_in[6];
    const float* fy        = (const float*)d_in[7];
    const float* cx        = (const float*)d_in[8];
    const float* cy        = (const float*)d_in[9];
    float* out = (float*)d_out;

    float4* pa = (float4*)d_ws;                                  // 2 MB
    v16h*   bt = (v16h*)((char*)d_ws + (size_t)2 * 1024 * 1024); // 2 MB

    prep_params<<<512, 256, 0, stream>>>(positions, scales, opac, qvec, tvec,
                                         fx, fy, cx, cy, pa);
    prep_btile<<<256, 256, 0, stream>>>(colors, bt);
    render_main<<<256, 128, 0, stream>>>(pa, bt, out);
}
